// FactorizedVectorQuantizer_38113539784837
// MI455X (gfx1250) — matrix operand order repaired and checked by reading, not compiled, not run on hardware
//
#include <hip/hip_runtime.h>
#include <stdint.h>

// ---------------------------------------------------------------------------
// FactorizedVectorQuantizer for MI455X (gfx1250, wave32, WMMA + TDM)
//   x:        [32, 256, 32, 32] fp32   (N = 32768 vectors of dim 256)
//   w_shape:  [8192, 128] fp32
//   w_color:  [ 512, 128] fp32
// Distances: ||w||^2 (exact fp32) - 2 * (flat . w) via bf16 WMMA (fp32 acc).
// Codebook tiles staged into LDS (shared by 8 waves/block) via the Tensor
// Data Mover (tensor_load_to_lds + s_wait_tensorcnt), double-buffered.
// ---------------------------------------------------------------------------

typedef __attribute__((ext_vector_type(16))) __bf16          v16bf;
typedef __attribute__((ext_vector_type(16))) unsigned short  u16x16;
typedef __attribute__((ext_vector_type(8)))  float           v8f;

struct U16Pack { unsigned short s[16]; };

__device__ __forceinline__ unsigned short f2bf(float f) {
  // round-to-nearest-even fp32 -> bf16 (inputs finite)
  unsigned int u = __builtin_bit_cast(unsigned int, f);
  u += 0x7FFFu + ((u >> 16) & 1u);
  return (unsigned short)(u >> 16);
}

#if defined(__has_builtin)
#if __has_builtin(__builtin_amdgcn_tensor_load_to_lds) && \
    __has_builtin(__builtin_amdgcn_s_wait_tensorcnt)
#define HAVE_TDM 1
#endif
#endif
#ifndef HAVE_TDM
#define HAVE_TDM 0
#endif

#if HAVE_TDM
typedef __attribute__((ext_vector_type(4))) unsigned int u32x4;
typedef __attribute__((ext_vector_type(8))) int          i32x8;
typedef __attribute__((ext_vector_type(4))) int          i32x4;

// 1-D contiguous global->LDS DMA of n8*8 bytes via the Tensor Data Mover.
// D# per CDNA5 ISA ch.8: group0 = {count=1, lds_addr, global_addr, type=2},
// group1 = {data_size=8B, tensor_dim0=tile_dim0=n8, tensor_dim1=tile_dim1=1}.
// This toolchain exposes the 6-arg builtin:
//   (uint32x4 g0, int32x8 g1, int32x4, int32x4, int32x8, i32 cpol)
__device__ __forceinline__ void tdm_load_1d(unsigned lds_off,
                                            const unsigned short* gptr,
                                            unsigned n8) {
  unsigned long long ga = (unsigned long long)(uintptr_t)gptr;
  u32x4 g0;
  g0.x = 1u;                                                // 1 valid D#
  g0.y = lds_off;                                           // lds_addr (bytes)
  g0.z = (unsigned)ga;                                      // global_addr[31:0]
  g0.w = (unsigned)((ga >> 32) & 0x1FFFFFFu) | (2u << 30);  // [56:32] | type=2
  i32x8 g1;
  g1[0] = (int)(3u << 16);                  // data_size=8B; wg_mask=0; no flags
  g1[1] = (int)((n8 & 0xFFFFu) << 16);      // tensor_dim0[15:0]  (bits 63:48)
  g1[2] = (int)((n8 >> 16) | (1u << 16));   // tensor_dim0[31:16] | tensor_dim1=1
  g1[3] = (int)((n8 & 0xFFFFu) << 16);      // tensor_dim1 hi | tile_dim0=n8
  g1[4] = 1;                                // tile_dim1=1, tile_dim2=0
  g1[5] = (int)n8;                          // tensor_dim0_stride[31:0]
  g1[6] = 0;
  g1[7] = 0;
  i32x4 z4 = {0, 0, 0, 0};
  i32x8 z8 = {0, 0, 0, 0, 0, 0, 0, 0};
  __builtin_amdgcn_tensor_load_to_lds(g0, g1, z4, z4, z8, 0);
}
#endif

// ---------------------------------------------------------------------------
// Pack codebook into bf16, pre-swizzled into WMMA B-matrix register layout:
//   layout index = ((chunk*4 + kc)*32 + lane)*16 + t
//   lane -> code column (lane & 15), lane half h: t<8 -> K 8h+t, t>=8 -> K 16+8h+(t-8) within kc
// ---------------------------------------------------------------------------
__global__ void __launch_bounds__(256)
pack_codebook(const float* __restrict__ w, unsigned short* __restrict__ wpk,
              int Kcodes) {
  int tid = blockIdx.x * blockDim.x + threadIdx.x;
  int total = Kcodes * 128;
  if (tid >= total) return;
  int t     = tid & 15;
  int lane  = (tid >> 4) & 31;
  int kc    = (tid >> 9) & 3;
  int chunk = tid >> 11;
  int code  = (chunk << 4) | (lane & 15);
  int d     = (kc << 5) + ((lane & 16) ? 8 : 0) + ((t < 8) ? t : t + 8);
  wpk[tid]  = f2bf(w[code * 128 + d]);
}

// ||w_k||^2 in exact fp32
__global__ void __launch_bounds__(256)
codebook_norms(const float* __restrict__ w, float* __restrict__ wnorm, int Kcodes) {
  int k = blockIdx.x * blockDim.x + threadIdx.x;
  if (k >= Kcodes) return;
  const float* row = w + (size_t)k * 128;
  float s = 0.f;
#pragma unroll 8
  for (int d = 0; d < 128; ++d) { float v = row[d]; s = fmaf(v, v, s); }
  wnorm[k] = s;
}

// ---------------------------------------------------------------------------
// Fused distance + argmin. Each of the 8 waves in a block owns a 16-row tile
// of flat; all waves sweep the same codes, whose packed bf16 B-tiles are
// staged into double-buffered LDS (32 KiB per stage = 8 chunks of 16 codes).
// flat[n][c] = x[b][c0+c][h][w], n = b*1024 + h*32 + w -> channel stride 1024.
// ---------------------------------------------------------------------------
__global__ void __launch_bounds__(256)
vq_argmin(const float* __restrict__ x, const unsigned short* __restrict__ wpk,
          const float* __restrict__ wnorm, int Kcodes, int c0,
          int* __restrict__ idx_out, int* __restrict__ counts) {
  __shared__ __align__(16) unsigned short sB[2][16384];   // 2 x 32 KiB

  const int lane  = threadIdx.x & 31;
  const int widx  = threadIdx.x >> 5;
  const int wave  = (blockIdx.x << 3) | widx;
  const int n0    = wave << 4;          // 16 rows per wave
  const int mrow  = lane & 15;
  const int row   = n0 + mrow;
  const int b     = row >> 10;
  const int hw    = row & 1023;
  const float* xrow = x + (((size_t)b * 256 + c0) << 10) + hw;
  const int base8 = (lane & 16) ? 8 : 0;

  // A tile (16 rows x 128 d), loaded once into WMMA A-register layout:
  // element t<8 -> K = base8+t ; t>=8 -> K = 16+base8+(t-8), per kc of 32.
  v16bf a[4];
#pragma unroll
  for (int kc = 0; kc < 4; ++kc) {
    U16Pack tmp;
#pragma unroll
    for (int t = 0; t < 8; ++t)
      tmp.s[t]     = f2bf(xrow[(size_t)((kc << 5) + base8 + t) << 10]);
#pragma unroll
    for (int t = 0; t < 8; ++t)
      tmp.s[8 + t] = f2bf(xrow[(size_t)((kc << 5) + 16 + base8 + t) << 10]);
    a[kc] = __builtin_bit_cast(v16bf, tmp);
  }

  float minv[8];
  int   mini[8];
#pragma unroll
  for (int r = 0; r < 8; ++r) { minv[r] = 3.4e38f; mini[r] = 0; }

  const int nchunk = Kcodes >> 4;
  const int nstage = nchunk >> 3;       // 8 chunks (32 KiB of B) per stage

#if HAVE_TDM
  if (widx == 0)
    tdm_load_1d((unsigned)(uintptr_t)&sB[0][0], wpk, 4096);
#endif

  for (int s = 0; s < nstage; ++s) {
    const int p = s & 1;

    // Hoist the 8 per-stage ||w||^2 values a lane needs (one clause, one wait)
    float wns[8];
#pragma unroll
    for (int j = 0; j < 8; ++j)
      wns[j] = wnorm[(s << 7) + (j << 4) + mrow];

#if HAVE_TDM
    if (widx == 0) {
      if (s + 1 < nstage) {
        tdm_load_1d((unsigned)(uintptr_t)&sB[p ^ 1][0],
                    wpk + ((size_t)(s + 1) << 14), 4096);
        __builtin_amdgcn_s_wait_tensorcnt(1);   // stage s done (TDM in-order)
      } else {
        __builtin_amdgcn_s_wait_tensorcnt(0);
      }
    }
    __syncthreads();                    // LDS stage visible to all waves
#else
    __syncthreads();                    // previous stage fully consumed
    {
      const u16x16* s16 = (const u16x16*)(wpk + ((size_t)s << 14));
      u16x16* d16 = (u16x16*)&sB[p][0];
      for (int i = threadIdx.x; i < 1024; i += 256) d16[i] = s16[i];
    }
    __syncthreads();
#endif

    // Two independent accumulator chains per unrolled pair: chunk j's argmin
    // VALU co-executes under chunk j+1's WMMAs instead of hazard NOPs.
#pragma unroll 2
    for (int j = 0; j < 8; ++j) {
      v8f acc = {0.f, 0.f, 0.f, 0.f, 0.f, 0.f, 0.f, 0.f};
      const u16x16* pB = (const u16x16*)&sB[p][(j << 11) + (lane << 4)];
#pragma unroll
      for (int kc = 0; kc < 4; ++kc) {
        v16bf bm = __builtin_bit_cast(v16bf, pB[kc * 32]);
        acc = __builtin_amdgcn_wmma_f32_16x16x32_bf16(
            /*neg_a=*/false, a[kc], /*neg_b=*/false, bm,
            /*c_mod=*/(short)0, acc, /*reuse_a=*/false, /*reuse_b=*/false);
      }
      const int   code = (((s << 3) + j) << 4) | mrow;
      const float wn   = wns[j];
#pragma unroll
      for (int r = 0; r < 8; ++r) {
        float d = fmaf(-2.f, acc[r], wn);     // ||w||^2 - 2 f.w
        if (d < minv[r]) { minv[r] = d; mini[r] = code; }
      }
    }
#if HAVE_TDM
    __syncthreads();                    // done reading buf p before overwrite
#endif
  }

  // Cross-lane argmin over the 16 code columns sharing each row.
  // C/D layout: lanes 0-15 hold rows 0-7 (per VGPR r), lanes 16-31 rows 8-15.
#pragma unroll
  for (int off = 8; off; off >>= 1) {
#pragma unroll
    for (int r = 0; r < 8; ++r) {
      float ov = __shfl_xor(minv[r], off, 32);
      int   oi = __shfl_xor(mini[r], off, 32);
      if (ov < minv[r] || (ov == minv[r] && oi < mini[r])) {
        minv[r] = ov; mini[r] = oi;
      }
    }
  }
  if (mrow == 0) {
    const int rb = (lane & 16) ? 8 : 0;
#pragma unroll
    for (int r = 0; r < 8; ++r) {
      idx_out[n0 + rb + r] = mini[r];
      atomicAdd(&counts[mini[r]], 1);
    }
  }
}

// ---------------------------------------------------------------------------
// Gather winning codes (exact fp32), write NCHW output, per-block SSE partial.
// ---------------------------------------------------------------------------
__global__ void __launch_bounds__(256)
vq_output(const float* __restrict__ x, const float* __restrict__ ws,
          const float* __restrict__ wc, const int* __restrict__ idx_s,
          const int* __restrict__ idx_c, float* __restrict__ out,
          float* __restrict__ blockSums) {
  __shared__ float lsum[8];
  size_t i = (size_t)blockIdx.x * 256 + threadIdx.x;  // linear NCHW index
  int hw = (int)(i & 1023);
  int c  = (int)((i >> 10) & 255);
  int b  = (int)(i >> 18);
  int n  = (b << 10) | hw;
  float q = (c < 128) ? ws[(size_t)idx_s[n] * 128 + c]
                      : wc[(size_t)idx_c[n] * 128 + (c - 128)];
  float f = x[i];
  out[i] = q;                      // straight-through output == q numerically
  float e = q - f; e *= e;
#pragma unroll
  for (int off = 16; off; off >>= 1) e += __shfl_xor(e, off, 32);
  if ((threadIdx.x & 31) == 0) lsum[threadIdx.x >> 5] = e;
  __syncthreads();
  if (threadIdx.x == 0) {
    float s = 0.f;
#pragma unroll
    for (int wv = 0; wv < 8; ++wv) s += lsum[wv];
    blockSums[blockIdx.x] = s;     // deterministic (no float atomics)
  }
}

// ---------------------------------------------------------------------------
// Finalize: loss = 1.25 * SSE / (N*C); perplexity = exp(-sum p log(p+eps)).
// ---------------------------------------------------------------------------
__global__ void __launch_bounds__(256)
vq_finalize(const float* __restrict__ blockSums, const int* __restrict__ cnt_s,
            const int* __restrict__ cnt_c, float* __restrict__ out3) {
  __shared__ float red[256];
  const int t = threadIdx.x;

  float s = 0.f;
  for (int i = t; i < 32768; i += 256) s += blockSums[i];
  red[t] = s; __syncthreads();
  for (int k = 128; k; k >>= 1) { if (t < k) red[t] += red[t + k]; __syncthreads(); }
  float loss = 1.25f * red[0] / (32768.f * 256.f);
  __syncthreads();

  float es = 0.f;
  for (int i = t; i < 8192; i += 256) {
    float p = (float)cnt_s[i] * (1.f / 32768.f);
    es += p * logf(p + 1e-10f);
  }
  red[t] = es; __syncthreads();
  for (int k = 128; k; k >>= 1) { if (t < k) red[t] += red[t + k]; __syncthreads(); }
  float perp_s = expf(-red[0]);
  __syncthreads();

  float ec = 0.f;
  for (int i = t; i < 512; i += 256) {
    float p = (float)cnt_c[i] * (1.f / 32768.f);
    ec += p * logf(p + 1e-10f);
  }
  red[t] = ec; __syncthreads();
  for (int k = 128; k; k >>= 1) { if (t < k) red[t] += red[t + k]; __syncthreads(); }

  if (t == 0) {
    out3[0] = loss;
    out3[1] = perp_s;
    out3[2] = expf(-red[0]);
  }
}

// ---------------------------------------------------------------------------
extern "C" void kernel_launch(void* const* d_in, const int* in_sizes, int n_in,
                              void* d_out, int out_size, void* d_ws, size_t ws_size,
                              hipStream_t stream) {
  (void)in_sizes; (void)n_in; (void)out_size; (void)ws_size;
  const float* x  = (const float*)d_in[0];   // [32,256,32,32]
  const float* ws = (const float*)d_in[1];   // [8192,128]
  const float* wc = (const float*)d_in[2];   // [512,128]
  float* out = (float*)d_out;                // 8388608 quantized + 3 scalars

  char* wsb = (char*)d_ws;                   // workspace layout (bytes)
  unsigned short* wpk_s = (unsigned short*)(wsb + 0);        // 2 MiB
  unsigned short* wpk_c = (unsigned short*)(wsb + 2097152);  // 128 KiB
  float* wn_s = (float*)(wsb + 2228224);                     // 32 KiB
  float* wn_c = (float*)(wsb + 2260992);                     // 2 KiB
  int*   idx_s = (int*)(wsb + 2263040);                      // 128 KiB
  int*   idx_c = (int*)(wsb + 2394112);                      // 128 KiB
  int*   cnt_s = (int*)(wsb + 2525184);                      // 32 KiB
  int*   cnt_c = (int*)(wsb + 2557952);                      // 2 KiB
  float* bsum  = (float*)(wsb + 2560000);                    // 128 KiB

  // zero the histograms (contiguous region); everything else is fully written
  (void)hipMemsetAsync(cnt_s, 0, 32768 + 2048, stream);

  pack_codebook<<<(8192 * 128) / 256, 256, 0, stream>>>(ws, wpk_s, 8192);
  pack_codebook<<<(512 * 128) / 256, 256, 0, stream>>>(wc, wpk_c, 512);
  codebook_norms<<<8192 / 256, 256, 0, stream>>>(ws, wn_s, 8192);
  codebook_norms<<<2, 256, 0, stream>>>(wc, wn_c, 512);

  // 2048 waves each (N/16 row-tiles), 8 waves per block sharing LDS B-tiles
  vq_argmin<<<256, 256, 0, stream>>>(x, wpk_s, wn_s, 8192, 0,   idx_s, cnt_s);
  vq_argmin<<<256, 256, 0, stream>>>(x, wpk_c, wn_c, 512,  128, idx_c, cnt_c);

  vq_output<<<32768, 256, 0, stream>>>(x, ws, wc, idx_s, idx_c, out, bsum);
  vq_finalize<<<1, 256, 0, stream>>>(bsum, cnt_s, cnt_c, out + 8388608);
}
